// TextCNN_88897233092836
// MI455X (gfx1250) — matrix operand order repaired and checked by reading, not compiled, not run on hardware
//
#include <hip/hip_runtime.h>
#include <hip/hip_bf16.h>

// ---------------------------------------------------------------------------
// TextCNN on MI455X (gfx1250): conv-as-GEMM with v_wmma_f32_16x16x32_bf16.
// e (B*S x 320 bf16) staged into LDS per K-chunk via the CDNA5 async
// global->LDS path (ASYNCcnt); Wz (64 x 33 x 320 bf16) streamed from L2.
// ---------------------------------------------------------------------------

typedef __attribute__((ext_vector_type(16))) __bf16 v16bf;
typedef __attribute__((ext_vector_type(8)))  float  v8f;
typedef __attribute__((ext_vector_type(4)))  unsigned int uint4v;
typedef int b128_t __attribute__((vector_size(16)));   // matches builtin param

#define BATCH   128
#define S_LEN   512
#define E_DIM   300
#define EPAD    320
#define NFILT   64
#define KW      33                 // max tap count (padded)
#define TILE_S  128                // output positions per workgroup
#define NSTILE  (S_LEN / TILE_S)   // 4
#define NPOS    160                // staged positions (TILE_S + 32 halo)
#define ROWV    5                  // uint4v per staged row (80 B padded stride)
#define AROWSZ  (KW * EPAD)        // elements per packed filter row (10560)

#if __has_builtin(__builtin_amdgcn_global_load_async_to_lds_b128)
#define HAVE_ASYNC_LDS 1
typedef __attribute__((address_space(1))) b128_t* gas_b128_p;
typedef __attribute__((address_space(3))) b128_t* las_b128_p;
#else
#define HAVE_ASYNC_LDS 0
#endif

// ---------------- Kernel 1: gather embeddings, convert to bf16, pad E ------
__global__ __launch_bounds__(160)
void tcnn_gather_kernel(const int* __restrict__ x,
                        const float* __restrict__ emb,
                        __bf16* __restrict__ eB)
{
    const int row = blockIdx.x;                 // b*512 + s
    const int token = x[row];
    const float* src = emb + (size_t)token * E_DIM;
    __bf16* dst = eB + (size_t)row * EPAD;
    for (int c = threadIdx.x; c < EPAD; c += 160) {
        float v = (c < E_DIM) ? src[c] : 0.0f;
        dst[c] = (__bf16)v;
    }
}

// ---------------- Kernel 2: pack/zero-pad conv weights to bf16 -------------
__global__ __launch_bounds__(160)
void tcnn_pack_kernel(const float* __restrict__ Wconv,
                      __bf16* __restrict__ Wz)
{
    const int ij = blockIdx.x;                  // i*33 + j
    const int i = ij / KW;
    const int j = ij % KW;
    const int ki = (i >> 1) + 2;
    const float* src = Wconv + (size_t)ij * E_DIM;
    __bf16* dst = Wz + (size_t)i * AROWSZ + (size_t)j * EPAD;
    for (int c = threadIdx.x; c < EPAD; c += 160) {
        float v = (j < ki && c < E_DIM) ? src[c] : 0.0f;
        dst[c] = (__bf16)v;
    }
}

// ---------------- Kernel 3: conv-GEMM via WMMA + tanh + masked max ---------
__global__ __launch_bounds__(256)
void tcnn_conv_wmma_kernel(const __bf16* __restrict__ eB,
                           const __bf16* __restrict__ Wz,
                           const float* __restrict__ bconv,
                           float* __restrict__ partial)   // [B][NSTILE][64]
{
    const int b    = blockIdx.y;
    const int st   = blockIdx.x;
    const int s0   = st * TILE_S;
    const int tid  = threadIdx.x;
    const int lane = tid & 31;
    const int lmod = lane & 15;
    const int half = lane >> 4;
    const int nt   = tid >> 5;            // one N-tile (16 positions) per wave

    __shared__ uint4v       sTile[NPOS * ROWV];   // 12,800 B staged e-chunk
    __shared__ unsigned int sMax[NFILT];
    if (tid < NFILT) sMax[tid] = 0u;      // encoded very-negative float

    v8f zero8 = {0.f,0.f,0.f,0.f,0.f,0.f,0.f,0.f};
    v8f acc[4];                           // all 4 M-tiles for this wave
    acc[0] = zero8; acc[1] = zero8; acc[2] = zero8; acc[3] = zero8;

    const __bf16* eBase = eB + (size_t)b * S_LEN * EPAD;

    for (int kc = 0; kc < EPAD; kc += 32) {
        __syncthreads();                  // protect readers of previous chunk
        // ---- stage 160 positions x 32 dims (bf16) -> LDS ----
        for (int idx = tid; idx < NPOS * 4; idx += 256) {
            const int row = idx >> 2;
            const int g   = idx & 3;
            const int pos = s0 + row;
            uint4v* dst = &sTile[row * ROWV + g];
            if (pos < S_LEN) {
                const __bf16* gp = eBase + (size_t)pos * EPAD + kc + g * 8;
#if HAVE_ASYNC_LDS
                // CDNA5 async DMA: cache -> LDS, no VGPR round-trip (ASYNCcnt)
                __builtin_amdgcn_global_load_async_to_lds_b128(
                    (gas_b128_p)gp, (las_b128_p)dst, 0, 0);
#else
                *dst = *(const uint4v*)gp;
#endif
            } else {
                uint4v z = {0u, 0u, 0u, 0u};
                *dst = z;                 // zero-fill halo beyond sequence end
            }
        }
#if HAVE_ASYNC_LDS
#if __has_builtin(__builtin_amdgcn_s_wait_asynccnt)
        __builtin_amdgcn_s_wait_asynccnt(0);
#endif
#endif
        __syncthreads();

        for (int j = 0; j < KW; ++j) {
            // --- B fragment (32x16 bf16) from LDS: contiguous 16-elem run ---
            const int pl = nt * 16 + lmod + j;           // 0..159
            const uint4v* bp = &sTile[pl * ROWV + half * 2];
            v16bf bfrag;
            ((uint4v*)&bfrag)[0] = bp[0];                // ds_load_b128
            ((uint4v*)&bfrag)[1] = bp[1];

            #pragma unroll
            for (int m = 0; m < 4; ++m) {
                if (j < 8 * m + 9) {      // widest filter in M-tile m: k=8m+9
                    // --- A fragment (16x32 bf16): two b128 runs, L2-hot ---
                    const __bf16* ap = Wz + (size_t)(m * 16 + lmod) * AROWSZ
                                          + j * EPAD + kc;
                    v16bf afrag;
                    ((uint4v*)&afrag)[0] = *(const uint4v*)(ap + half * 16);
                    ((uint4v*)&afrag)[1] = *(const uint4v*)(ap + half * 16 + 8);
                    acc[m] = __builtin_amdgcn_wmma_f32_16x16x32_bf16(
                        false, afrag, false, bfrag, (short)0, acc[m],
                        false, false);
                }
            }
        }
    }

    // ---- Epilogue: bias + tanh + validity mask + max reduce ----
    const int s = s0 + nt * 16 + lmod;    // this thread's position
    #pragma unroll
    for (int m = 0; m < 4; ++m) {
        #pragma unroll
        for (int r = 0; r < 8; ++r) {
            const int f  = m * 16 + half * 8 + r;   // C/D layout: M=r (+8 upper)
            const int kf = (f >> 1) + 2;
            const int Lf = S_LEN - kf + 1;
            const float tv = tanhf(acc[m][r] + bconv[f]);
            float vmax = (s < Lf) ? tv : -2.0f;     // tanh in (-1,1)
            // butterfly max across the 16 lanes sharing filter f
            #pragma unroll
            for (int d = 1; d < 16; d <<= 1)
                vmax = fmaxf(vmax, __shfl_xor(vmax, d, 32));
            if (lmod == 0) {
                unsigned int bits = __float_as_uint(vmax);
                unsigned int enc  = (bits & 0x80000000u) ? ~bits
                                                         : (bits | 0x80000000u);
                atomicMax(&sMax[f], enc);
            }
        }
    }
    __syncthreads();

    if (tid < NFILT) {
        const unsigned int u = sMax[tid];
        const unsigned int bits = (u & 0x80000000u) ? (u ^ 0x80000000u) : ~u;
        partial[((size_t)b * NSTILE + st) * NFILT + tid] = __uint_as_float(bits);
    }
}

// ---------------- Kernel 4: reduce tiles, linear + sigmoid -----------------
__global__ __launch_bounds__(64)
void tcnn_finalize_kernel(const float* __restrict__ partial,
                          const float* __restrict__ Wlin,
                          const float* __restrict__ blin,
                          float* __restrict__ out)
{
    const int b = blockIdx.x;
    const int f = threadIdx.x;        // 64 threads
    float mx = -2.0f;
    for (int st = 0; st < NSTILE; ++st)
        mx = fmaxf(mx, partial[((size_t)b * NSTILE + st) * NFILT + f]);
    __shared__ float red[NFILT];
    red[f] = mx * Wlin[f];
    __syncthreads();
    if (f < 32) red[f] += red[f + 32];
    __syncthreads();
    if (f == 0) {
        float s = 0.0f;
        for (int i = 0; i < 32; ++i) s += red[i];
        out[b] = 1.0f / (1.0f + expf(-(s + blin[0])));
    }
}

// ---------------------------------------------------------------------------
extern "C" void kernel_launch(void* const* d_in, const int* in_sizes, int n_in,
                              void* d_out, int out_size, void* d_ws, size_t ws_size,
                              hipStream_t stream)
{
    const int*   x     = (const int*)  d_in[0];   // (B,1,S) int
    const float* emb   = (const float*)d_in[1];   // (VOCAB,E)
    const float* Wconv = (const float*)d_in[2];   // (H,KMAX,E)
    const float* bconv = (const float*)d_in[3];   // (H,)
    const float* Wlin  = (const float*)d_in[4];   // (1,H)
    const float* blin  = (const float*)d_in[5];   // (1,)
    float* out = (float*)d_out;                   // (B,1) f32

    char* ws = (char*)d_ws;
    size_t off = 0;
    __bf16* eB = (__bf16*)(ws + off);
    off += (size_t)BATCH * S_LEN * EPAD * sizeof(__bf16);   // 41,943,040 B
    __bf16* Wz = (__bf16*)(ws + off);
    off += (size_t)NFILT * AROWSZ * sizeof(__bf16);         // 1,351,680 B
    off = (off + 255) & ~(size_t)255;
    float* partial = (float*)(ws + off);                    // 131,072 B

    tcnn_gather_kernel<<<BATCH * S_LEN, 160, 0, stream>>>(x, emb, eB);
    tcnn_pack_kernel<<<NFILT * KW, 160, 0, stream>>>(Wconv, Wz);
    dim3 grid(NSTILE, BATCH);
    tcnn_conv_wmma_kernel<<<grid, 256, 0, stream>>>(eB, Wz, bconv, partial);
    tcnn_finalize_kernel<<<BATCH, NFILT, 0, stream>>>(partial, Wlin, blin, out);
}
